// TextDecoder_20409684590894
// MI455X (gfx1250) — matrix operand order repaired and checked by reading, not compiled, not run on hardware
//
#include <hip/hip_runtime.h>
#include <hip/hip_bf16.h>
#include <math.h>

// ---------------------------------------------------------------------------
// LSTM text decoder: V=512, L=128, B=512.
//   gates = onehot(idx) @ W_ih^T + b_ih + h @ W_hh^T + b_hh   (order i,f,g,o)
// One-hot matmul = row gather from pre-transposed W_ih. Recurrence GEMM
// (512x2048x512 per step, 127 steps) runs on v_wmma_f32_16x16x32_bf16 with
// f32 accumulation; pointwise math in f32. Each wave owns TWO 16-row M-tiles
// so every weight (B) fragment feeds two WMMAs: 12 b128 loads per 8 WMMAs.
// ---------------------------------------------------------------------------

typedef __attribute__((ext_vector_type(16))) __bf16 v16bf;
typedef __attribute__((ext_vector_type(8)))  __bf16 v8bf;
typedef __attribute__((ext_vector_type(8)))  float  v8f;

#define VOCAB 512
#define GATES 2048          // 4*V
#define BATCH 512
#define SEQL  128

// ---------------- workspace layout (bytes) ---------------------------------
#define WS_WHH_BF16 (0u)                               // 2048*512*2  = 2 MiB
#define WS_WIHT     (WS_WHH_BF16 + 2097152u)           // 512*2048*4  = 4 MiB
#define WS_BSUM     (WS_WIHT + 4194304u)               // 2048*4
#define WS_HB0      (WS_BSUM + 8192u)                  // 512*512*2
#define WS_HB1      (WS_HB0 + 524288u)                 // 512*512*2
#define WS_HF32     (WS_HB1 + 524288u)                 // 512*512*4
#define WS_CBUF     (WS_HF32 + 1048576u)               // 512*512*4
#define WS_IDX      (WS_CBUF + 1048576u)               // 512*4
// total ~9.45 MiB

__device__ __forceinline__ float sigmoidf_(float x) {
    return 1.0f / (1.0f + __expf(-x));
}

__device__ __forceinline__ v16bf cat8(v8bf a, v8bf b) {
    return __builtin_shufflevector(a, b,
        0,1,2,3,4,5,6,7,8,9,10,11,12,13,14,15);
}

// ---------------------------------------------------------------------------
// Prologue: weight conversion/transpose, bias fold, state init, PAD row.
// ---------------------------------------------------------------------------
__global__ void decoder_prologue(const float* __restrict__ x,
                                 const float* __restrict__ W_ih,
                                 const float* __restrict__ W_hh,
                                 const float* __restrict__ b_ih,
                                 const float* __restrict__ b_hh,
                                 __bf16* __restrict__ Whh_bf16,
                                 float*  __restrict__ WihT,
                                 float*  __restrict__ bsum,
                                 __bf16* __restrict__ h0_bf16,
                                 float*  __restrict__ cbuf,
                                 int*    __restrict__ idx,
                                 float*  __restrict__ out) {
    int i = blockIdx.x * blockDim.x + threadIdx.x;      // 0 .. 2048*512-1
    if (i < GATES * VOCAB) {
        float w = W_hh[i];
        Whh_bf16[i] = (__bf16)w;
        int j = i >> 9;           // gate row 0..2047
        int v = i & 511;          // vocab col
        WihT[(size_t)v * GATES + j] = W_ih[i];          // transpose for gather
    }
    if (i < BATCH * VOCAB) {
        cbuf[i] = 0.0f;
        h0_bf16[i] = (__bf16)x[i];                      // h0 = x
        int b = i >> 9, v = i & 511;
        // last sequence position stays PAD one-hot
        out[(size_t)b * SEQL * VOCAB + (size_t)(SEQL - 1) * VOCAB + v] =
            (v == 0) ? 1.0f : 0.0f;
    }
    if (i < GATES) bsum[i] = b_ih[i] + b_hh[i];
    if (i < BATCH) idx[i] = 1;                          // SOS_ID
}

// ---------------------------------------------------------------------------
// Fused gate GEMM + LSTM cell update.
// grid = (V/16, B/128), block = 128 (4 waves). Wave w: rows m_base..+31
// (two 16-row M-tiles), cols jb..jb+15 of V, 4 gates -> 8 accumulators.
// Each B (weight) fragment is reused by two WMMAs.
// ---------------------------------------------------------------------------
__global__ __launch_bounds__(128)
void lstm_step_wmma(const __bf16* __restrict__ h_in,      // [B,V] bf16 (read)
                    const __bf16* __restrict__ Whh,       // [4V,V] bf16
                    const float*  __restrict__ WihT,      // [V,4V] f32
                    const float*  __restrict__ bsum,      // [4V]
                    const int*    __restrict__ idx,       // [B]
                    float*        __restrict__ cbuf,      // [B,V] in-place
                    float*        __restrict__ h_out_f32, // [B,V]
                    __bf16*       __restrict__ h_out_bf16)// [B,V] (write buf)
{
    const int wave = threadIdx.x >> 5;
    const int lane = threadIdx.x & 31;
    const int lo   = lane & 15;
    const int hi   = lane >> 4;

    const int m_base = blockIdx.y * 128 + wave * 32;    // two 16-row tiles
    const int jb     = blockIdx.x * 16;

    v8f acc00 = {}; v8f acc10 = {}; v8f acc20 = {}; v8f acc30 = {}; // tile 0
    v8f acc01 = {}; v8f acc11 = {}; v8f acc21 = {}; v8f acc31 = {}; // tile 1

    // A rows: h rows (m_base+lo) and (m_base+16+lo). B rows: W_hh row
    // (gate*512 + jb + lo). K is contiguous in memory for both.
    const __bf16* arow0 = h_in + (size_t)(m_base + lo) * VOCAB;
    const __bf16* arow1 = h_in + (size_t)(m_base + 16 + lo) * VOCAB;
    const __bf16* brow0 = Whh + (size_t)(0 * VOCAB + jb + lo) * VOCAB;
    const __bf16* brow1 = Whh + (size_t)(1 * VOCAB + jb + lo) * VOCAB;
    const __bf16* brow2 = Whh + (size_t)(2 * VOCAB + jb + lo) * VOCAB;
    const __bf16* brow3 = Whh + (size_t)(3 * VOCAB + jb + lo) * VOCAB;

    #pragma unroll 2
    for (int kk = 0; kk < VOCAB; kk += 32) {
        // A fragments: lanes 0-15 hold K kk..kk+15,
        //              lanes 16-31 hold K kk+16..kk+31
        v16bf A0 = cat8(*(const v8bf*)(arow0 + kk + hi * 16),
                        *(const v8bf*)(arow0 + kk + hi * 16 + 8));
        v16bf A1 = cat8(*(const v8bf*)(arow1 + kk + hi * 16),
                        *(const v8bf*)(arow1 + kk + hi * 16 + 8));

        // B fragments: lanes 0-15 K=kk..kk+15, lanes 16-31 K=kk+16..kk+31
        {
            v16bf B = cat8(*(const v8bf*)(brow0 + kk + hi * 16),
                           *(const v8bf*)(brow0 + kk + hi * 16 + 8));
            acc00 = __builtin_amdgcn_wmma_f32_16x16x32_bf16(
                false, A0, false, B, (short)0, acc00, false, false);
            acc01 = __builtin_amdgcn_wmma_f32_16x16x32_bf16(
                false, A1, false, B, (short)0, acc01, false, false);
        }
        {
            v16bf B = cat8(*(const v8bf*)(brow1 + kk + hi * 16),
                           *(const v8bf*)(brow1 + kk + hi * 16 + 8));
            acc10 = __builtin_amdgcn_wmma_f32_16x16x32_bf16(
                false, A0, false, B, (short)0, acc10, false, false);
            acc11 = __builtin_amdgcn_wmma_f32_16x16x32_bf16(
                false, A1, false, B, (short)0, acc11, false, false);
        }
        {
            v16bf B = cat8(*(const v8bf*)(brow2 + kk + hi * 16),
                           *(const v8bf*)(brow2 + kk + hi * 16 + 8));
            acc20 = __builtin_amdgcn_wmma_f32_16x16x32_bf16(
                false, A0, false, B, (short)0, acc20, false, false);
            acc21 = __builtin_amdgcn_wmma_f32_16x16x32_bf16(
                false, A1, false, B, (short)0, acc21, false, false);
        }
        {
            v16bf B = cat8(*(const v8bf*)(brow3 + kk + hi * 16),
                           *(const v8bf*)(brow3 + kk + hi * 16 + 8));
            acc30 = __builtin_amdgcn_wmma_f32_16x16x32_bf16(
                false, A0, false, B, (short)0, acc30, false, false);
            acc31 = __builtin_amdgcn_wmma_f32_16x16x32_bf16(
                false, A1, false, B, (short)0, acc31, false, false);
        }
    }

    // D layout: VGPR r holds (M = r + hi*8, N = lo). Fused LSTM pointwise.
    const int jcol = jb + lo;
    const float bi = bsum[0 * VOCAB + jcol];
    const float bf = bsum[1 * VOCAB + jcol];
    const float bg = bsum[2 * VOCAB + jcol];
    const float bo = bsum[3 * VOCAB + jcol];

    #pragma unroll
    for (int tile = 0; tile < 2; ++tile) {
        const v8f a0 = tile ? acc01 : acc00;
        const v8f a1 = tile ? acc11 : acc10;
        const v8f a2 = tile ? acc21 : acc20;
        const v8f a3 = tile ? acc31 : acc30;
        const int mt = m_base + tile * 16;
        #pragma unroll
        for (int r = 0; r < 8; ++r) {
            const int m = mt + hi * 8 + r;
            const size_t e = (size_t)m * VOCAB + jcol;
            const float* wih = WihT + (size_t)idx[m] * GATES + jcol;

            float iv = a0[r] + bi + wih[0 * VOCAB];
            float fv = a1[r] + bf + wih[1 * VOCAB];
            float gv = a2[r] + bg + wih[2 * VOCAB];
            float ov = a3[r] + bo + wih[3 * VOCAB];

            float c_old = cbuf[e];
            float c_new = sigmoidf_(fv) * c_old + sigmoidf_(iv) * tanhf(gv);
            float h_new = sigmoidf_(ov) * tanhf(c_new);

            cbuf[e]       = c_new;
            h_out_f32[e]  = h_new;
            h_out_bf16[e] = (__bf16)h_new;
        }
    }
}

// ---------------------------------------------------------------------------
// Softmax over V=512 + first-occurrence argmax -> probas + next token.
// grid = B, block = 256 (each thread covers 2 vocab entries).
// ---------------------------------------------------------------------------
__global__ __launch_bounds__(256)
void softmax_argmax(const float* __restrict__ h,    // [B,V]
                    float* __restrict__ out,        // [B,L,V]
                    int*   __restrict__ idx,        // [B]
                    int t) {
    __shared__ float sval[256];
    __shared__ int   sarg[256];
    __shared__ float ssum[256];

    const int b   = blockIdx.x;
    const int tid = threadIdx.x;
    const float* row = h + (size_t)b * VOCAB;

    float v0 = row[tid];
    float v1 = row[tid + 256];
    float mv = v0; int ma = tid;
    if (v1 > mv) { mv = v1; ma = tid + 256; }   // strict: keeps lower index
    sval[tid] = mv; sarg[tid] = ma;
    __syncthreads();
    for (int s = 128; s > 0; s >>= 1) {
        if (tid < s) {
            float ov = sval[tid + s]; int oa = sarg[tid + s];
            if (ov > sval[tid] || (ov == sval[tid] && oa < sarg[tid])) {
                sval[tid] = ov; sarg[tid] = oa;
            }
        }
        __syncthreads();
    }
    const float rmax = sval[0];
    if (tid == 0) idx[b] = sarg[0];

    float e0 = __expf(v0 - rmax);
    float e1 = __expf(v1 - rmax);
    ssum[tid] = e0 + e1;
    __syncthreads();
    for (int s = 128; s > 0; s >>= 1) {
        if (tid < s) ssum[tid] += ssum[tid + s];
        __syncthreads();
    }
    const float inv = 1.0f / ssum[0];

    float* orow = out + (size_t)b * SEQL * VOCAB + (size_t)t * VOCAB;
    orow[tid]       = e0 * inv;
    orow[tid + 256] = e1 * inv;
}

// ---------------------------------------------------------------------------
extern "C" void kernel_launch(void* const* d_in, const int* in_sizes, int n_in,
                              void* d_out, int out_size, void* d_ws, size_t ws_size,
                              hipStream_t stream) {
    const float* x    = (const float*)d_in[0];
    const float* W_ih = (const float*)d_in[1];
    const float* W_hh = (const float*)d_in[2];
    const float* b_ih = (const float*)d_in[3];
    const float* b_hh = (const float*)d_in[4];
    float* out = (float*)d_out;

    char* ws = (char*)d_ws;
    __bf16* Whh_bf16 = (__bf16*)(ws + WS_WHH_BF16);
    float*  WihT     = (float*) (ws + WS_WIHT);
    float*  bsum     = (float*) (ws + WS_BSUM);
    __bf16* hb0      = (__bf16*)(ws + WS_HB0);
    __bf16* hb1      = (__bf16*)(ws + WS_HB1);
    float*  hf32     = (float*) (ws + WS_HF32);
    float*  cbuf     = (float*) (ws + WS_CBUF);
    int*    idxp     = (int*)   (ws + WS_IDX);

    decoder_prologue<<<(GATES * VOCAB + 255) / 256, 256, 0, stream>>>(
        x, W_ih, W_hh, b_ih, b_hh,
        Whh_bf16, WihT, bsum, hb0, cbuf, idxp, out);

    dim3 gA(VOCAB / 16, BATCH / 128);   // (32, 4)
    for (int t = 0; t < SEQL - 1; ++t) {
        const __bf16* hin  = (t & 1) ? hb1 : hb0;
        __bf16*       hout = (t & 1) ? hb0 : hb1;
        lstm_step_wmma<<<gA, 128, 0, stream>>>(
            hin, Whh_bf16, WihT, bsum, idxp, cbuf, hf32, hout);
        softmax_argmax<<<BATCH, 256, 0, stream>>>(hf32, out, idxp, t);
    }
}
